// OrderedFeatureGroupANFIS_80985903333981
// MI455X (gfx1250) — compile-verified, matrix operand order checked by reading, not run on hardware
//
#include <hip/hip_runtime.h>
#include <hip/hip_bf16.h>

typedef __bf16 bf16_t;
typedef __attribute__((ext_vector_type(16))) __bf16 v16bf;
typedef __attribute__((ext_vector_type(8)))  float  v8f;
typedef __attribute__((ext_vector_type(4)))  unsigned int v4u;
typedef __attribute__((ext_vector_type(8)))  int v8i;
typedef __attribute__((ext_vector_type(4)))  int v4i;

constexpr int BATCH  = 4096;
constexpr int NFEAT  = 12;
constexpr int RULES  = 4096;   // 2^12, rule r selects MF (r>>f)&1 for feature f
constexpr int OUT    = 8;
constexpr int NPAD   = 112;    // 7 x 16 N-tiles (96 sugeno + 8 bias + 8 zero)
constexpr int NTILE  = 7;
constexpr int NKC    = RULES / 32;        // 128 K-chunks of 32
constexpr int BLKROW = 64;                // batch rows per block (4 waves x 16)
constexpr int CHUNK_ELE = NTILE * 32 * 16;   // 3584 bf16 per K-chunk, fragment-ordered
constexpr int CHUNK_BYTES = CHUNK_ELE * 2;   // 7168 bytes

__device__ __forceinline__ float softplus(float z) {
    return (z > 20.f) ? z : log1pf(expf(z));
}

// ---------------------------------------------------------------------------
// Kernel 1: per-batch-row feature-normalized memberships + global scale.
//   mhat[b,f,j] = m_j / (m0+m1)   (0.5 guard when m0+m1 underflows to 0)
//   scale[b]    = S / (S + 1e-8),  S = prod_f (m0+m1)
// firing_norm[b,r] = scale[b] * prod_f mhat[b,f,(r>>f)&1]
// ---------------------------------------------------------------------------
__global__ void k_prep_memb(const float* __restrict__ x,
                            const float* __restrict__ cbase,
                            const float* __restrict__ cdelta,
                            const float* __restrict__ wraw,
                            float* __restrict__ mhat,   // [BATCH,12,2]
                            float* __restrict__ scale)  // [BATCH]
{
    int b = blockIdx.x * blockDim.x + threadIdx.x;
    if (b >= BATCH) return;
    float S = 1.f;
    #pragma unroll
    for (int f = 0; f < NFEAT; ++f) {
        float c0 = cbase[f];
        float c1 = c0 + softplus(cdelta[f]) + 1e-3f;
        float w0 = softplus(wraw[f * 2 + 0]) + 1e-3f;
        float w1 = softplus(wraw[f * 2 + 1]) + 1e-3f;
        float xv = x[b * NFEAT + f];
        float d0 = xv - c0, d1 = xv - c1;
        float m0 = expf(-(d0 * d0) / (2.f * w0 * w0));
        float m1 = expf(-(d1 * d1) / (2.f * w1 * w1));
        float t  = m0 + m1;
        float h0, h1;
        if (t > 0.f) { h0 = m0 / t; h1 = m1 / t; } else { h0 = 0.5f; h1 = 0.5f; }
        mhat[(b * NFEAT + f) * 2 + 0] = h0;
        mhat[(b * NFEAT + f) * 2 + 1] = h1;
        S *= t;
    }
    scale[b] = S / (S + 1e-8f);
}

// ---------------------------------------------------------------------------
// Kernel 2: pack B-panel as bf16, PRE-SWIZZLED into WMMA B-fragment order and
// chunk-major, so each K-chunk is one contiguous 7168-byte blob (TDM-friendly).
// Logical col n: 0..95 = p[r, f=n/8, o=n%8], 96..103 = consequent_r, rest 0.
// Fragment order: k -> (lanehi=(k>>3)&1, elem=(k&7)|((k>>4)<<3)), n -> (t, n&15)
// ---------------------------------------------------------------------------
__global__ void k_prep_bpanel(const float* __restrict__ p,
                              const float* __restrict__ cr,
                              bf16_t* __restrict__ Bg)
{
    int idx = blockIdx.x * blockDim.x + threadIdx.x;
    if (idx >= RULES * NPAD) return;
    int r = idx / NPAD, n = idx % NPAD;
    float v = 0.f;
    if (n < 96)        v = p[r * 96 + n];
    else if (n < 104)  v = cr[r * OUT + (n - 96)];
    int kc = r >> 5, k = r & 31;
    int t = n >> 4;
    int l = (n & 15) | (((k >> 3) & 1) << 4);
    int e = (k & 7) | ((k >> 4) << 3);
    Bg[kc * CHUNK_ELE + ((t * 32 + l) << 4) + e] = (bf16_t)v;
}

// ---------------------------------------------------------------------------
// TDM: 1-D copy of one 7168-byte chunk into LDS via Tensor Data Mover.
// D# per CDNA5 ISA ch.8: group0 {count=1, lds_addr, global_addr, type=2},
// group1 {data_size=3 (8B), tensor_dim0=tile_dim0=896, dim1=1, strides=896}.
// Toolchain here is the clang-23 lane: 6-arg builtin
//   (v4u g0, v8i g1, v4i g2, v4i g3, v8i g4, i32 cpol)
// ---------------------------------------------------------------------------
#if __has_builtin(__builtin_amdgcn_tensor_load_to_lds)
#define HAVE_TDM 1
__device__ __forceinline__ void tdm_load_chunk(const bf16_t* gsrc, unsigned lds_addr)
{
    unsigned long long ga = (unsigned long long)(uintptr_t)gsrc;
    v4u g0 = { 1u,                                    // count=1, user descriptor
               lds_addr,                              // LDS byte address
               (unsigned)(ga & 0xFFFFFFFFu),          // global_addr[31:0]
               (unsigned)((ga >> 32) & 0x1FFFFFFu) | (2u << 30) }; // [56:32] | type=2
    v8i g1 = { (int)(3u << 16),          // data_size=3 -> 8-byte elements
               (int)(896u << 16),        // tensor_dim0[15:0]=896 (bits 63:48)
               (int)(1u << 16),          // tensor_dim0 hi=0 | tensor_dim1 lo=1
               (int)(896u << 16),        // tensor_dim1 hi=0 | tile_dim0=896
               (int)(1u),                // tile_dim1=1, tile_dim2=0
               (int)(896u),              // tensor_dim0_stride[31:0]=896
               (int)(896u << 16),        // stride hi=0 | tensor_dim1_stride lo=896
               0 };                      // tensor_dim1_stride hi=0
    v4i g2 = { 0, 0, 0, 0 };             // dims >=3 unused
    v4i g3 = { 0, 0, 0, 0 };
    v8i g4 = { 0, 0, 0, 0, 0, 0, 0, 0 }; // trailing group unused (<=2D, no gather)
    __builtin_amdgcn_tensor_load_to_lds(g0, g1, g2, g3, g4, 0);
}
#else
#define HAVE_TDM 0
#endif

// ---------------------------------------------------------------------------
// Kernel 3: WMMA GEMM  G[BATCH, NPAD] = Ahat[BATCH, RULES] x Bg[RULES, NPAD]
// A tiles synthesized on the fly: Ahat[b,r] = Plo[b][r&63] * Phi[b][r>>6];
// B chunks DMA'd by the TDM into a double-buffered LDS stage.
// ---------------------------------------------------------------------------
__global__ __launch_bounds__(128)
void k_gemm(const float* __restrict__ mhat,
            const bf16_t* __restrict__ Bg,
            float* __restrict__ G)
{
    __shared__ float lds_mh[BLKROW * NFEAT * 2];            // 6 KB
    __shared__ float Plo[BLKROW * 65];                      // stride 65: conflict-free
    __shared__ float Phi[BLKROW * 65];
    __shared__ __align__(32) bf16_t ldsB[2][CHUNK_ELE];     // 2 x 7 KB double buffer

    const int tid  = threadIdx.x;
    const int lane = tid & 31;
    const int wave = tid >> 5;
    const int rowbase = blockIdx.x * BLKROW;

    // stage this block's mhat rows
    for (int i = tid; i < BLKROW * 24; i += 128)
        lds_mh[i] = mhat[rowbase * 24 + i];

#if HAVE_TDM
    if (tid < 32)                                   // wave 0 owns the TDM pipe
        tdm_load_chunk(Bg, (unsigned)(uintptr_t)&ldsB[0][0]);
#endif
    __syncthreads();

    // build per-row partial-product tables
    for (int i = tid; i < BLKROW * 64; i += 128) {
        int bl = i >> 6, c = i & 63;
        const float* mr = &lds_mh[bl * 24];
        float plo = 1.f, phi = 1.f;
        #pragma unroll
        for (int f = 0; f < 6; ++f) {
            int bit = (c >> f) & 1;
            plo *= mr[f * 2 + bit];
            phi *= mr[(f + 6) * 2 + bit];
        }
        Plo[bl * 65 + c] = plo;
        Phi[bl * 65 + c] = phi;
    }

    v8f acc[NTILE];
    v8f zero = {0.f, 0.f, 0.f, 0.f, 0.f, 0.f, 0.f, 0.f};
    #pragma unroll
    for (int t = 0; t < NTILE; ++t) acc[t] = zero;

    const int lanelo = lane & 15;
    const int khalf  = (lane >> 4) << 3;          // ISA 16-bit A layout: upper lanes K+8
    const float* PloRow = &Plo[(wave * 16 + lanelo) * 65];
    const float* PhiRow = &Phi[(wave * 16 + lanelo) * 65];
    const unsigned ldsB0 = (unsigned)(uintptr_t)&ldsB[0][0];

    for (int kc = 0; kc < NKC; ++kc) {
        const bf16_t* buf = ldsB[kc & 1];

#if HAVE_TDM
        if (tid < 32) __builtin_amdgcn_s_wait_tensorcnt(0);   // chunk kc landed
        __syncthreads();                                      // all waves see it; all done
                                                              // reading buf[(kc+1)&1]
        if (tid < 32 && kc + 1 < NKC)                         // overlap DMA with compute
            tdm_load_chunk(Bg + (kc + 1) * CHUNK_ELE,
                           ldsB0 + ((kc + 1) & 1) * CHUNK_BYTES);
#else
        __syncthreads();
        {   // fallback: plain contiguous cooperative copy (panel is pre-swizzled)
            const unsigned int* src = (const unsigned int*)(Bg + kc * CHUNK_ELE);
            unsigned int* dst = (unsigned int*)buf;
            for (int i = tid; i < CHUNK_ELE / 2; i += 128) dst[i] = src[i];
        }
        __syncthreads();
#endif

        // synthesize A fragment (16 bf16 per lane, ISA 16-bit A layout)
        v16bf afrag;
        #pragma unroll
        for (int i = 0; i < 16; ++i) {
            int klocal = khalf + (i < 8 ? i : i + 8);
            int kg = kc * 32 + klocal;
            afrag[i] = (bf16_t)(PloRow[kg & 63] * PhiRow[kg >> 6]);
        }

        // hoist all 7 B fragments, then issue the 7 WMMAs back-to-back
        v16bf bfr[NTILE];
        #pragma unroll
        for (int t = 0; t < NTILE; ++t)
            bfr[t] = *(const v16bf*)&buf[(t * 32 + lane) << 4];
        #pragma unroll
        for (int t = 0; t < NTILE; ++t)
            acc[t] = __builtin_amdgcn_wmma_f32_16x16x32_bf16(
                false, afrag, false, bfr[t], (short)0, acc[t], false, false);

#if HAVE_TDM
        __syncthreads();   // readers of buf[kc&1] done before it is refilled (kc+2)
#endif
    }

    // epilogue: C layout -> row = v + (lane>=16 ? 8 : 0), col = lane&15
    const int mofs = (lane >> 4) << 3;
    #pragma unroll
    for (int t = 0; t < NTILE; ++t) {
        #pragma unroll
        for (int v = 0; v < 8; ++v) {
            int b = rowbase + wave * 16 + mofs + v;
            int n = t * 16 + lanelo;
            G[b * NPAD + n] = acc[t][v];
        }
    }
}

// ---------------------------------------------------------------------------
// Kernel 4: out[b,o] = scale[b] * ( G[b,96+o] + sum_f x[b,f]*G[b,f*8+o] )
// ---------------------------------------------------------------------------
__global__ void k_final(const float* __restrict__ x,
                        const float* __restrict__ G,
                        const float* __restrict__ scale,
                        float* __restrict__ out)
{
    int idx = blockIdx.x * blockDim.x + threadIdx.x;
    if (idx >= BATCH * OUT) return;
    int b = idx >> 3, o = idx & 7;
    float a = G[b * NPAD + 96 + o];
    #pragma unroll
    for (int f = 0; f < NFEAT; ++f)
        a += x[b * NFEAT + f] * G[b * NPAD + f * 8 + o];
    out[idx] = scale[b] * a;
}

// ---------------------------------------------------------------------------
extern "C" void kernel_launch(void* const* d_in, const int* in_sizes, int n_in,
                              void* d_out, int out_size, void* d_ws, size_t ws_size,
                              hipStream_t stream) {
    const float* x      = (const float*)d_in[0];
    const float* cbase  = (const float*)d_in[1];
    const float* cdelta = (const float*)d_in[2];
    const float* wraw   = (const float*)d_in[3];
    const float* p      = (const float*)d_in[4];
    const float* cr     = (const float*)d_in[5];
    float* out = (float*)d_out;

    char* ws = (char*)d_ws;
    float*  mhat  = (float*)(ws + 0);                        // 4096*24*4  = 393216
    float*  scale = (float*)(ws + 393216);                   // 4096*4     = 16384
    bf16_t* Bg    = (bf16_t*)(ws + 409600);                  // 4096*112*2 = 917504
    float*  G     = (float*)(ws + 1327104);                  // 4096*112*4 = 1835008
    // total ws use: ~3.02 MB

    k_prep_memb<<<(BATCH + 255) / 256, 256, 0, stream>>>(x, cbase, cdelta, wraw, mhat, scale);
    k_prep_bpanel<<<(RULES * NPAD + 255) / 256, 256, 0, stream>>>(p, cr, Bg);
    k_gemm<<<BATCH / BLKROW, 128, 0, stream>>>(mhat, Bg, G);
    k_final<<<(BATCH * OUT + 255) / 256, 256, 0, stream>>>(x, G, scale, out);
}
